// CausalSelfAttention_15496242004497
// MI455X (gfx1250) — matrix operand order repaired and checked by reading, not compiled, not run on hardware
//
#include <hip/hip_runtime.h>

typedef unsigned short u16;
typedef __attribute__((ext_vector_type(16))) __bf16 bf16x16;
typedef __attribute__((ext_vector_type(8)))  float  f32x8;
typedef int v4i __attribute__((vector_size(16)));   // matches builtin param type

union Frag {
    bf16x16 v;
    uint4   u[2];
};

__device__ __forceinline__ u16 f2bf(float f) {
    unsigned int u = __float_as_uint(f);
    unsigned int r = (u + 0x7FFFu + ((u >> 16) & 1u)) >> 16;   // RNE
    return (u16)r;
}

// ---- CDNA5 async global->LDS staging (guarded; falls back to plain copy) --
#if defined(__has_builtin)
#if __has_builtin(__builtin_amdgcn_global_load_async_to_lds_b128)
#define HAVE_ASYNC_LDS 1
#endif
#endif
#ifndef HAVE_ASYNC_LDS
#define HAVE_ASYNC_LDS 0
#endif

__device__ __forceinline__ void stageB32(u16* dst, const u16* src) {
#if HAVE_ASYNC_LDS
    __builtin_amdgcn_global_load_async_to_lds_b128((v4i*)src, (v4i*)dst, 0, 0);
    __builtin_amdgcn_global_load_async_to_lds_b128((v4i*)(src + 8), (v4i*)(dst + 8), 0, 0);
#else
    *(uint4*)dst       = *(const uint4*)src;
    *(uint4*)(dst + 8) = *(const uint4*)(src + 8);
#endif
}

__device__ __forceinline__ void wait_async() {
#if HAVE_ASYNC_LDS
#if __has_builtin(__builtin_amdgcn_s_wait_asynccnt)
    __builtin_amdgcn_s_wait_asynccnt(0);
#else
    asm volatile("s_wait_asynccnt 0" ::: "memory");
#endif
#endif
}

// ---------------------------------------------------------------- convert
__global__ void cvt_bf16(const float* __restrict__ in, u16* __restrict__ out, int n4) {
    int i = blockIdx.x * blockDim.x + threadIdx.x;
    if (i < n4) {
        float4 f = ((const float4*)in)[i];
        union { u16 s[4]; uint2 u; } r;
        r.s[0] = f2bf(f.x); r.s[1] = f2bf(f.y);
        r.s[2] = f2bf(f.z); r.s[3] = f2bf(f.w);
        ((uint2*)out)[i] = r.u;
    }
}

// ---------------------------------------------------------------- GEMM
// y = A[8192,1024] @ W[1024,1024]^T.
// Block tile 128x64: 4 waves x (32x64), B tile shared via double-buffered LDS
// staged with GLOBAL_LOAD_ASYNC_TO_LDS (ASYNCcnt) one K-step ahead.
// MODE 0: store bf16 as [B,H,T,64]   (Q and K)
// MODE 2: store bf16 as [B,H,64,T]   (V transposed)
// MODE 3: store f32  as [8192,1024]  (final output)
template <int MODE>
__global__ void __launch_bounds__(128) gemm128x64(const u16* __restrict__ A,
                                                  const u16* __restrict__ W,
                                                  void* __restrict__ out) {
    __shared__ u16 Bs[2][64 * 32];          // 2 x 4KB double buffer

    const int tid  = threadIdx.x;
    const int wid  = tid >> 5;
    const int lane = tid & 31;
    const int hf   = lane >> 4;
    const int lm   = lane & 15;
    const int c0   = hf * 8;

    const int nBase = (blockIdx.x & 15) * 64;             // 16 n-tiles
    const int mBase = (blockIdx.x >> 4) * 128 + wid * 32; // 64 m-tiles of 128

    // cooperative B staging: thread copies 32B of the 64x32 tile
    const int crow   = tid >> 1;            // 0..63
    const int cchunk = (tid & 1) * 16;      // u16 offset 0 / 16
    const u16* wsrc = W + (size_t)(nBase + crow) * 1024 + cchunk;
    u16* bdst0 = &Bs[0][crow * 32 + cchunk];
    u16* bdst1 = &Bs[1][crow * 32 + cchunk];

    const u16* arow0 = A + (size_t)(mBase + lm) * 1024;
    const u16* arow1 = A + (size_t)(mBase + 16 + lm) * 1024;

    f32x8 acc[2][4] = {};

    stageB32(bdst0, wsrc);                  // prefetch k = 0
    for (int ki = 0; ki < 32; ++ki) {
        const int k = ki * 32;
        wait_async();
        __syncthreads();                    // buffer ki&1 resident everywhere
        if (ki + 1 < 32)
            stageB32((ki & 1) ? bdst0 : bdst1, wsrc + (ki + 1) * 32);

        Frag a0, a1;
        a0.u[0] = *(const uint4*)(arow0 + k + c0);
        a0.u[1] = *(const uint4*)(arow0 + k + 16 + c0);
        a1.u[0] = *(const uint4*)(arow1 + k + c0);
        a1.u[1] = *(const uint4*)(arow1 + k + 16 + c0);

        const u16* bs = Bs[ki & 1];
#pragma unroll
        for (int nt = 0; nt < 4; ++nt) {
            Frag bf;
            const u16* bp = bs + (nt * 16 + lm) * 32 + hf * 8;
            bf.u[0] = *(const uint4*)(bp);
            bf.u[1] = *(const uint4*)(bp + 16);
            acc[0][nt] = __builtin_amdgcn_wmma_f32_16x16x32_bf16(
                false, a0.v, false, bf.v, (short)0, acc[0][nt], false, false);
            acc[1][nt] = __builtin_amdgcn_wmma_f32_16x16x32_bf16(
                false, a1.v, false, bf.v, (short)0, acc[1][nt], false, false);
        }
    }

#pragma unroll
    for (int mf = 0; mf < 2; ++mf) {
#pragma unroll
        for (int nt = 0; nt < 4; ++nt) {
#pragma unroll
            for (int r = 0; r < 8; ++r) {
                const int m = mBase + mf * 16 + r + 8 * hf;
                const int n = nBase + nt * 16 + lm;
                const float val = acc[mf][nt][r];
                if (MODE == 3) {
                    ((float*)out)[(size_t)m * 1024 + n] = val;
                } else {
                    const int bb = m >> 11, t = m & 2047;     // T = 2048
                    const int hh = n >> 6,  dd = n & 63;      // Dh = 64
                    if (MODE == 2)
                        ((u16*)out)[(((size_t)bb * 16 + hh) * 64 + dd) * 2048 + t] = f2bf(val);
                    else
                        ((u16*)out)[(((size_t)bb * 16 + hh) * 2048 + t) * 64 + dd] = f2bf(val);
                }
            }
        }
    }
}

// ---------------------------------------------------------------- attention
// Flash-attention: each wave = 16 queries x Dh=64, key tiles of 32.
// Q,K: [B,H,T,64] bf16.  Vt: [B,H,64,T] bf16.  attnb: [B,T,1024] bf16.
__global__ void __launch_bounds__(128) attn_kernel(const u16* __restrict__ Qb,
                                                   const u16* __restrict__ Kb,
                                                   const u16* __restrict__ Vt,
                                                   u16* __restrict__ attnb) {
    __shared__ u16 pS[4][16 * 32];

    const int tid  = threadIdx.x;
    const int wid  = tid >> 5;
    const int lane = tid & 31;
    const int hf   = lane >> 4;
    const int lm   = lane & 15;
    const int c0   = hf * 8;

    const int bid   = blockIdx.x;        // 64 (b,h) * 32 q64-tiles
    const int bh    = bid >> 5;          // b*16 + h
    const int qBase = (bid & 31) * 64 + wid * 16;

    const u16* Qp = Qb + (size_t)bh * 2048 * 64;
    const u16* Kp = Kb + (size_t)bh * 2048 * 64;
    const u16* Vp = Vt + (size_t)bh * 64 * 2048;

    Frag qa[2];
#pragma unroll
    for (int ks = 0; ks < 2; ++ks) {
        const u16* p = Qp + (size_t)(qBase + lm) * 64 + ks * 32;
        qa[ks].u[0] = *(const uint4*)(p + c0);
        qa[ks].u[1] = *(const uint4*)(p + 16 + c0);
    }

    f32x8 o[4] = {};
    float mrow[8], lrow[8];
#pragma unroll
    for (int r = 0; r < 8; ++r) { mrow[r] = -3.0e38f; lrow[r] = 0.0f; }

    const int ktCount = qBase / 32 + 1;          // causal bound, 32-key tiles
    for (int kt = 0; kt < ktCount; ++kt) {
        const int kBase = kt * 32;

        f32x8 s0 = {}, s1 = {};
#pragma unroll
        for (int ks = 0; ks < 2; ++ks) {
            Frag kf0, kf1;
            const u16* pk0 = Kp + (size_t)(kBase + lm) * 64 + ks * 32 + hf * 8;
            kf0.u[0] = *(const uint4*)(pk0);
            kf0.u[1] = *(const uint4*)(pk0 + 16);
            const u16* pk1 = Kp + (size_t)(kBase + 16 + lm) * 64 + ks * 32 + hf * 8;
            kf1.u[0] = *(const uint4*)(pk1);
            kf1.u[1] = *(const uint4*)(pk1 + 16);
            s0 = __builtin_amdgcn_wmma_f32_16x16x32_bf16(
                false, qa[ks].v, false, kf0.v, (short)0, s0, false, false);
            s1 = __builtin_amdgcn_wmma_f32_16x16x32_bf16(
                false, qa[ks].v, false, kf1.v, (short)0, s1, false, false);
        }

        float p0[8], p1[8];
#pragma unroll
        for (int r = 0; r < 8; ++r) {
            const int q = qBase + r + 8 * hf;
            float a = s0[r] * 0.125f;                       // 1/sqrt(64)
            if (kBase + lm > q)       a = -3.0e38f;
            float b = s1[r] * 0.125f;
            if (kBase + 16 + lm > q)  b = -3.0e38f;

            float rv = fmaxf(a, b);
#pragma unroll
            for (int off = 1; off < 16; off <<= 1)
                rv = fmaxf(rv, __shfl_xor(rv, off, 32));
            const float mn = fmaxf(mrow[r], rv);
            const float al = __expf(mrow[r] - mn);
            const float e0 = __expf(a - mn);
            const float e1 = __expf(b - mn);
            float rs = e0 + e1;
#pragma unroll
            for (int off = 1; off < 16; off <<= 1)
                rs += __shfl_xor(rs, off, 32);
            lrow[r] = lrow[r] * al + rs;
            mrow[r] = mn;
            o[0][r] *= al; o[1][r] *= al; o[2][r] *= al; o[3][r] *= al;
            p0[r] = e0; p1[r] = e1;
        }

        // C-layout -> A-fragment via LDS (same-wave DS ops are in-order)
#pragma unroll
        for (int r = 0; r < 8; ++r) {
            const int row = r + 8 * hf;
            pS[wid][row * 32 + lm]      = f2bf(p0[r]);
            pS[wid][row * 32 + 16 + lm] = f2bf(p1[r]);
        }
        Frag pa;
        pa.u[0] = *(const uint4*)&pS[wid][lm * 32 + c0];
        pa.u[1] = *(const uint4*)&pS[wid][lm * 32 + 16 + c0];

        // O += P(16x32) . V(32x64); Vt layout keeps B-operand contiguous
#pragma unroll
        for (int t = 0; t < 4; ++t) {
            Frag vf;
            const u16* pv = Vp + (size_t)(t * 16 + lm) * 2048 + kBase + hf * 8;
            vf.u[0] = *(const uint4*)(pv);
            vf.u[1] = *(const uint4*)(pv + 16);
            o[t] = __builtin_amdgcn_wmma_f32_16x16x32_bf16(
                false, pa.v, false, vf.v, (short)0, o[t], false, false);
        }
    }

#pragma unroll
    for (int r = 0; r < 8; ++r) {
        const int q = qBase + r + 8 * hf;
        const float inv = 1.0f / lrow[r];
        const size_t rowOff = ((size_t)(bh >> 4) * 2048 + q) * 1024 + (size_t)(bh & 15) * 64;
#pragma unroll
        for (int t = 0; t < 4; ++t)
            attnb[rowOff + t * 16 + lm] = f2bf(o[t][r] * inv);
    }
}

// ---------------------------------------------------------------- launch
extern "C" void kernel_launch(void* const* d_in, const int* in_sizes, int n_in,
                              void* d_out, int out_size, void* d_ws, size_t ws_size,
                              hipStream_t stream) {
    const float* x  = (const float*)d_in[0];
    const float* wq = (const float*)d_in[1];
    const float* wk = (const float*)d_in[2];
    const float* wv = (const float*)d_in[3];
    const float* wo = (const float*)d_in[4];

    const size_t NX = (size_t)8192 * 1024;   // B*T x DIM
    const size_t NW = (size_t)1024 * 1024;

    u16* xb    = (u16*)d_ws;
    u16* wqb   = xb + NX;
    u16* wkb   = wqb + NW;
    u16* wvb   = wkb + NW;
    u16* wob   = wvb + NW;
    u16* Qb    = wob + NW;
    u16* Kb    = Qb + NX;
    u16* Vtb   = Kb + NX;
    u16* attnb = Vtb + NX;

    cvt_bf16<<<(int)(NX / 4 / 256), 256, 0, stream>>>(x, xb, (int)(NX / 4));
    cvt_bf16<<<(int)(NW / 4 / 256), 256, 0, stream>>>(wq, wqb, (int)(NW / 4));
    cvt_bf16<<<(int)(NW / 4 / 256), 256, 0, stream>>>(wk, wkb, (int)(NW / 4));
    cvt_bf16<<<(int)(NW / 4 / 256), 256, 0, stream>>>(wv, wvb, (int)(NW / 4));
    cvt_bf16<<<(int)(NW / 4 / 256), 256, 0, stream>>>(wo, wob, (int)(NW / 4));

    // 64 m-tiles x 16 n-tiles = 1024 blocks, 128 threads (4 waves)
    gemm128x64<0><<<1024, 128, 0, stream>>>(xb, wqb, Qb);
    gemm128x64<0><<<1024, 128, 0, stream>>>(xb, wkb, Kb);
    gemm128x64<2><<<1024, 128, 0, stream>>>(xb, wvb, Vtb);

    // 64 (b,h) pairs * 32 query-64 tiles = 2048 blocks, 4 waves each
    attn_kernel<<<2048, 128, 0, stream>>>(Qb, Kb, Vtb, attnb);

    gemm128x64<3><<<1024, 128, 0, stream>>>(attnb, wob, d_out);
}
